// CausalSelfAttention_16234976379501
// MI455X (gfx1250) — matrix operand order repaired and checked by reading, not compiled, not run on hardware
//
#include <hip/hip_runtime.h>
#include <stdint.h>

#define B_  4
#define T_  2048
#define C_  1024
#define H_  16
#define HS_ 64
static constexpr float ATTN_SCALE = 0.125f; // 1/sqrt(64)

typedef __attribute__((ext_vector_type(16))) __bf16    v16bf;
typedef __attribute__((ext_vector_type(8)))  float     v8f;
typedef __attribute__((ext_vector_type(4)))  uint32_t  v4u;

// A/B fragment: 16 bf16 per lane = 2 x 16-byte chunks.
union Frag16 { v16bf v; v4u q[2]; uint32_t u[8]; };
union BPack  { v4u q[2]; uint16_t e[16]; };   // 16 halfwords (one B column slice)
union VPack  { v4u q[4]; uint16_t e[32]; };   // 32 halfwords (one V column slice)

__device__ __forceinline__ uint16_t f32_to_bf16(float f) {
  union { float f; uint32_t u; } c; c.f = f;
  uint32_t r = c.u + 0x7FFFu + ((c.u >> 16) & 1u);
  return (uint16_t)(r >> 16);
}

__device__ __forceinline__ v8f v8f_zero() {
  v8f z;
#pragma unroll
  for (int i = 0; i < 8; ++i) z[i] = 0.f;
  return z;
}

// Generic pointer to LDS: low 32 bits are the LDS byte offset (ISA 10.2).
__device__ __forceinline__ uint32_t lds_off(const void* p) {
  return (uint32_t)(uintptr_t)p;
}

// CDNA5 async global->LDS copy, 16B per lane; tracked with ASYNCcnt.
__device__ __forceinline__ void async_copy_b128(uint32_t lds_addr, const void* gaddr) {
  asm volatile("global_load_async_to_lds_b128 %0, %1, off"
               :: "v"(lds_addr), "v"(gaddr) : "memory");
}
__device__ __forceinline__ void async_wait0() {
  asm volatile("s_wait_asynccnt 0x0" ::: "memory");
}

// ---------------------------------------------------------------------------
// f32 -> bf16 conversion
// ---------------------------------------------------------------------------
__global__ void f32_to_bf16_kernel(const float* __restrict__ src,
                                   uint16_t* __restrict__ dst, int n) {
  int i = blockIdx.x * blockDim.x + threadIdx.x;
  int stride = gridDim.x * blockDim.x;
  for (; i < n; i += stride) dst[i] = f32_to_bf16(src[i]);
}

// ---------------------------------------------------------------------------
// bf16 WMMA GEMM: out = A(MxK) @ B(KxN) + bias(N)
// Block tile 128x128, 256 threads = 8 waves (4x2), wave tile 32x64 (2x4 WMMA).
// Double-buffered LDS; next tile staged async while current tile computes.
// LDS strides are 40 halfwords: 16B-aligned rows for conflict-light b128 ops.
// ---------------------------------------------------------------------------
template <bool OUT_BF16>
__global__ __launch_bounds__(256)
void gemm_bf16_wmma(const uint16_t* __restrict__ A,
                    const uint16_t* __restrict__ Bm,
                    const float* __restrict__ bias,
                    uint16_t* __restrict__ outb,
                    float* __restrict__ outf,
                    int M, int N, int K) {
  __shared__ uint16_t At[2][128][40];   // row-major A tiles (128 x 32)
  __shared__ uint16_t Bt[2][128][40];   // transposed B tiles: Bt[n][k]

  const int t    = threadIdx.x;
  const int lane = t & 31;
  const int wave = t >> 5;
  const int wm   = wave >> 1;        // 0..3  (M direction, 32 rows each)
  const int wn   = wave & 1;         // 0..1  (N direction, 64 cols each)
  const int half = lane >> 4;
  const int ln16 = lane & 15;

  const int n0 = blockIdx.x * 128;
  const int r0 = blockIdx.y * 128;

  const int arow = t >> 1;           // 0..127
  const int ac0  = (t & 1) * 16;     // 0 or 16
  const int bn   = t & 127;          // B column this thread stages
  const int kh   = (t >> 7) * 16;    // 0 or 16: K half this thread stages

  // --- staging helpers -----------------------------------------------------
  auto stage_A_async = [&](int kk, int buf) {
    const uint16_t* asrc = A + (size_t)(r0 + arow) * K + kk + ac0;
    async_copy_b128(lds_off(&At[buf][arow][ac0]),     asrc);
    async_copy_b128(lds_off(&At[buf][arow][ac0 + 8]), asrc + 8);
  };
  auto b_load = [&](int kk) {        // lane-coalesced strided reads of column bn
    BPack p;
#pragma unroll
    for (int i = 0; i < 16; ++i)
      p.e[i] = Bm[(size_t)(kk + kh + i) * N + n0 + bn];
    return p;
  };
  auto b_store = [&](int buf, const BPack& p) {   // 2 x ds_store_b128
    *reinterpret_cast<v4u*>(&Bt[buf][bn][kh])     = p.q[0];
    *reinterpret_cast<v4u*>(&Bt[buf][bn][kh + 8]) = p.q[1];
  };

  v8f acc[2][4];
#pragma unroll
  for (int mi = 0; mi < 2; ++mi)
#pragma unroll
    for (int nj = 0; nj < 4; ++nj) acc[mi][nj] = v8f_zero();

  // --- prologue: stage first tile ------------------------------------------
  stage_A_async(0, 0);
  { BPack p0 = b_load(0); b_store(0, p0); }
  async_wait0();
  __syncthreads();

  int cur = 0;
  for (int kk = 0; kk < K; kk += 32) {
    const int  nxt       = cur ^ 1;
    const bool have_next = (kk + 32 < K);
    BPack bp;
    if (have_next) {                 // issue next-tile traffic before compute
      stage_A_async(kk + 32, nxt);
      bp = b_load(kk + 32);
    }

    // --- fragments from current buffers: 2 x ds_load_b128 each
    Frag16 af[2], bf[4];
#pragma unroll
    for (int mi = 0; mi < 2; ++mi) {
      const uint16_t* ar = &At[cur][wm * 32 + mi * 16 + ln16][0];
      af[mi].q[0] = *reinterpret_cast<const v4u*>(ar + half * 8);
      af[mi].q[1] = *reinterpret_cast<const v4u*>(ar + 16 + half * 8);
    }
#pragma unroll
    for (int nj = 0; nj < 4; ++nj) {
      const uint16_t* br = &Bt[cur][wn * 64 + nj * 16 + ln16][0];
      bf[nj].q[0] = *reinterpret_cast<const v4u*>(br + half * 8);
      bf[nj].q[1] = *reinterpret_cast<const v4u*>(br + 16 + half * 8);
    }
#pragma unroll
    for (int mi = 0; mi < 2; ++mi)
#pragma unroll
      for (int nj = 0; nj < 4; ++nj)
        acc[mi][nj] = __builtin_amdgcn_wmma_f32_16x16x32_bf16(
            false, af[mi].v, false, bf[nj].v, (short)0, acc[mi][nj], false, false);

    if (have_next) b_store(nxt, bp); // B-load wait lands after the WMMAs
    async_wait0();
    __syncthreads();
    cur = nxt;
  }

  // --- epilogue: bias + store
  const int mbase = half * 8;
#pragma unroll
  for (int nj = 0; nj < 4; ++nj) {
    int col  = n0 + wn * 64 + nj * 16 + ln16;
    float bv = bias[col];
#pragma unroll
    for (int mi = 0; mi < 2; ++mi) {
#pragma unroll
      for (int r = 0; r < 8; ++r) {
        int row = r0 + wm * 32 + mi * 16 + mbase + r;
        float v = acc[mi][nj][r] + bv;
        if (OUT_BF16) outb[(size_t)row * N + col] = f32_to_bf16(v);
        else          outf[(size_t)row * N + col] = v;
      }
    }
  }
}

// ---------------------------------------------------------------------------
// Flash attention: one block = 64 query rows of one (b,h); 4 waves x 16 rows.
// Double-buffered K/V tiles; next tile's async copies stay in flight across
// the softmax barriers (barriers do not drain ASYNCcnt).
// qkv layout: [(b*T+t)*3C + {0:q,C:k,2C:v} + h*64 + d], bf16.
// y layout:   [(b*T+t)*C + h*64 + d], bf16.
// ---------------------------------------------------------------------------
__global__ __launch_bounds__(128)
void flash_attn_wmma(const uint16_t* __restrict__ qkv,
                     uint16_t* __restrict__ y) {
  __shared__ uint16_t Kt[2][64][72];   // K tiles (key x d) == B^T form for QK^T
  __shared__ uint16_t Vt[2][64][72];   // V tiles transposed (d x key) for P@V
  __shared__ float    Sbuf[4][16][64]; // per-wave raw scores
  __shared__ uint16_t Pbuf[4][16][72]; // per-wave probabilities (bf16)
  __shared__ float    Rbuf[4][16];     // per-row accumulator rescale
  __shared__ float    Lbuf[4][16];     // per-row 1/l

  const int t     = threadIdx.x;
  const int lane  = t & 31;
  const int wave  = t >> 5;
  const int half  = lane >> 4;
  const int ln16  = lane & 15;
  const int mbase = half * 8;

  const int bh = blockIdx.y;
  const int b  = bh / H_;
  const int h  = bh % H_;
  const int q0 = blockIdx.x * 64;
  const int qw = q0 + wave * 16;       // this wave's first query row

  const int kkey = t >> 1;             // K-staging: key row, 2 threads per row
  const int kd0  = (t & 1) * 32;       //            d half
  const int vd   = t & 63;             // V-staging: d column this thread owns
  const int vkh  = (t >> 6) * 32;      //            key half

  auto stage_K_async = [&](int it, int buf) {
    const uint16_t* src =
        qkv + (size_t)(b * T_ + it * 64 + kkey) * (3 * C_) + C_ + h * HS_ + kd0;
#pragma unroll
    for (int i = 0; i < 4; ++i)
      async_copy_b128(lds_off(&Kt[buf][kkey][kd0 + 8 * i]), src + 8 * i);
  };
  auto v_load = [&](int it) {          // lane-coalesced strided reads of column vd
    VPack p;
#pragma unroll
    for (int i = 0; i < 32; ++i)
      p.e[i] = qkv[(size_t)(b * T_ + it * 64 + vkh + i) * (3 * C_) + 2 * C_ +
                   h * HS_ + vd];
    return p;
  };
  auto v_store = [&](int buf, const VPack& p) {   // 4 x ds_store_b128
#pragma unroll
    for (int i = 0; i < 4; ++i)
      *reinterpret_cast<v4u*>(&Vt[buf][vd][vkh + 8 * i]) = p.q[i];
  };

  // Q A-frags: 4 x global b128 loads, kept in registers for the whole kernel
  Frag16 qa[2];
  {
    const uint16_t* qrow =
        qkv + (size_t)(b * T_ + qw + ln16) * (3 * C_) + h * HS_;
#pragma unroll
    for (int ks = 0; ks < 2; ++ks) {
      qa[ks].q[0] = *reinterpret_cast<const v4u*>(qrow + ks * 32 + half * 8);
      qa[ks].q[1] = *reinterpret_cast<const v4u*>(qrow + ks * 32 + 16 + half * 8);
    }
  }

  v8f o[4];
#pragma unroll
  for (int j = 0; j < 4; ++j) o[j] = v8f_zero();
  float m_state = -__builtin_inff();
  float l_state = 0.f;

  const int ntiles = blockIdx.x + 1;   // causal: only tiles up to the diagonal

  // --- prologue: stage tile 0
  stage_K_async(0, 0);
  { VPack p0 = v_load(0); v_store(0, p0); }
  async_wait0();
  __syncthreads();

  int cur = 0;
  for (int it = 0; it < ntiles; ++it) {
    const int  kt0       = it * 64;
    const int  nxt       = cur ^ 1;
    const bool have_next = (it + 1 < ntiles);
    VPack vp;
    if (have_next) {                   // next tile in flight during this tile
      stage_K_async(it + 1, nxt);
      vp = v_load(it + 1);
    }

    // --- S = Q @ K^T  (4 key ntiles x 2 k-steps over d)
    v8f s[4];
#pragma unroll
    for (int j = 0; j < 4; ++j) s[j] = v8f_zero();
#pragma unroll
    for (int j = 0; j < 4; ++j) {
#pragma unroll
      for (int ks = 0; ks < 2; ++ks) {
        Frag16 kb;
        const uint16_t* kr = &Kt[cur][j * 16 + ln16][0];
        kb.q[0] = *reinterpret_cast<const v4u*>(kr + ks * 32 + half * 8);
        kb.q[1] = *reinterpret_cast<const v4u*>(kr + ks * 32 + 16 + half * 8);
        s[j] = __builtin_amdgcn_wmma_f32_16x16x32_bf16(
            false, qa[ks].v, false, kb.v, (short)0, s[j], false, false);
      }
    }

    // --- spill S to LDS for row-wise softmax
#pragma unroll
    for (int j = 0; j < 4; ++j)
#pragma unroll
      for (int r = 0; r < 8; ++r)
        Sbuf[wave][mbase + r][j * 16 + ln16] = s[j][r];
    __syncthreads();

    // --- online softmax: lanes 0..15 of each wave each own one query row
    if (lane < 16) {
      const int row = lane;
      const int q   = qw + row;
      float m_new = m_state;
      for (int c = 0; c < 64; ++c)
        if (kt0 + c <= q)
          m_new = fmaxf(m_new, Sbuf[wave][row][c] * ATTN_SCALE);
      float lad = 0.f;
      for (int c = 0; c < 64; ++c) {
        float p = 0.f;
        if (kt0 + c <= q) {
          p = __expf(Sbuf[wave][row][c] * ATTN_SCALE - m_new);
          lad += p;
        }
        Pbuf[wave][row][c] = f32_to_bf16(p);
      }
      float rs = __expf(m_state - m_new);
      l_state  = l_state * rs + lad;
      m_state  = m_new;
      Rbuf[wave][row] = rs;
    }
    __syncthreads();

    // --- rescale running O by per-row factor
#pragma unroll
    for (int r = 0; r < 8; ++r) {
      float rs = Rbuf[wave][mbase + r];
#pragma unroll
      for (int j = 0; j < 4; ++j) o[j][r] *= rs;
    }

    // --- O += P @ V  (contract over 64 keys: 2 k-steps; 4 d ntiles)
#pragma unroll
    for (int ks = 0; ks < 2; ++ks) {
      Frag16 pa;
      const uint16_t* pr = &Pbuf[wave][ln16][0];
      pa.q[0] = *reinterpret_cast<const v4u*>(pr + ks * 32 + half * 8);
      pa.q[1] = *reinterpret_cast<const v4u*>(pr + ks * 32 + 16 + half * 8);
#pragma unroll
      for (int j = 0; j < 4; ++j) {
        Frag16 vb;
        const uint16_t* vr = &Vt[cur][j * 16 + ln16][0];
        vb.q[0] = *reinterpret_cast<const v4u*>(vr + ks * 32 + half * 8);
        vb.q[1] = *reinterpret_cast<const v4u*>(vr + ks * 32 + 16 + half * 8);
        o[j] = __builtin_amdgcn_wmma_f32_16x16x32_bf16(
            false, pa.v, false, vb.v, (short)0, o[j], false, false);
      }
    }

    if (have_next) v_store(nxt, vp);   // V-load wait lands after the WMMAs
    async_wait0();
    __syncthreads();
    cur = nxt;
  }

  // --- final normalization and store
  if (lane < 16) Lbuf[wave][lane] = 1.0f / l_state;
  __syncthreads();
#pragma unroll
  for (int r = 0; r < 8; ++r) {
    const float li = Lbuf[wave][mbase + r];
    const int   q  = qw + mbase + r;
#pragma unroll
    for (int j = 0; j < 4; ++j) {
      int d = j * 16 + ln16;
      y[(size_t)(b * T_ + q) * C_ + h * HS_ + d] = f32_to_bf16(o[j][r] * li);
    }
  }
}

// ---------------------------------------------------------------------------
// Launcher
// ---------------------------------------------------------------------------
extern "C" void kernel_launch(void* const* d_in, const int* in_sizes, int n_in,
                              void* d_out, int out_size, void* d_ws, size_t ws_size,
                              hipStream_t stream) {
  (void)in_sizes; (void)n_in; (void)out_size; (void)ws_size;
  const float* x  = (const float*)d_in[0];
  const float* w1 = (const float*)d_in[1];   // c_attn_w (C x 3C)
  const float* b1 = (const float*)d_in[2];   // c_attn_b (3C)
  const float* w2 = (const float*)d_in[3];   // c_proj_w (C x C)
  const float* b2 = (const float*)d_in[4];   // c_proj_b (C)
  float* out = (float*)d_out;

  const int M = B_ * T_;                     // 8192 rows

  char* ws = (char*)d_ws;
  size_t off = 0;
  uint16_t* x_bf   = (uint16_t*)(ws + off); off += (size_t)M * C_ * 2;
  uint16_t* w1_bf  = (uint16_t*)(ws + off); off += (size_t)C_ * 3 * C_ * 2;
  uint16_t* w2_bf  = (uint16_t*)(ws + off); off += (size_t)C_ * C_ * 2;
  uint16_t* qkv_bf = (uint16_t*)(ws + off); off += (size_t)M * 3 * C_ * 2;
  uint16_t* y_bf   = (uint16_t*)(ws + off);

  f32_to_bf16_kernel<<<2048, 256, 0, stream>>>(x,  x_bf,  M * C_);
  f32_to_bf16_kernel<<<2048, 256, 0, stream>>>(w1, w1_bf, C_ * 3 * C_);
  f32_to_bf16_kernel<<<1024, 256, 0, stream>>>(w2, w2_bf, C_ * C_);

  // qkv = x @ c_attn_w + c_attn_b  -> bf16
  gemm_bf16_wmma<true><<<dim3(3 * C_ / 128, M / 128), 256, 0, stream>>>(
      x_bf, w1_bf, b1, qkv_bf, nullptr, M, 3 * C_, C_);

  // flash attention -> y_bf
  flash_attn_wmma<<<dim3(T_ / 64, B_ * H_), 128, 0, stream>>>(qkv_bf, y_bf);

  // out = y @ c_proj_w + c_proj_b -> f32
  gemm_bf16_wmma<false><<<dim3(C_ / 128, M / 128), 256, 0, stream>>>(
      y_bf, w2_bf, b2, nullptr, out, M, C_, C_);
}
